// QMLP_swin_46729244180915
// MI455X (gfx1250) — matrix operand order repaired and checked by reading, not compiled, not run on hardware
//
#include <hip/hip_runtime.h>

typedef _Float16 f16;
typedef __attribute__((ext_vector_type(16))) _Float16 v16h;
typedef __attribute__((ext_vector_type(8)))  _Float16 v8h;
typedef __attribute__((ext_vector_type(4)))  _Float16 v4h;
typedef __attribute__((ext_vector_type(8)))  float    v8f;
typedef __attribute__((ext_vector_type(4)))  float    v4f;

#define SQRT127 11.269427669584644f

// ---------------------------------------------------------------------------
// CDNA5 async global->LDS copy (GVS mode), tracked by ASYNCcnt.
// ---------------------------------------------------------------------------
__device__ __forceinline__ void async_copy_b128(unsigned lds_byte,
                                                const void* gbase,
                                                unsigned goff) {
  asm volatile("global_load_async_to_lds_b128 %0, %1, %2"
               :: "v"(lds_byte), "v"(goff), "s"(gbase)
               : "memory");
}

// ---------------------------------------------------------------------------
// Stage 1: activation fake-quant codes: q = clamp(round(x / s_a1[c]), -128, 127)
// ---------------------------------------------------------------------------
__global__ __launch_bounds__(256) void k_quant_act(
    const float* __restrict__ x, const float* __restrict__ s_a1,
    f16* __restrict__ aq, int total4, int C) {
  int idx = blockIdx.x * blockDim.x + threadIdx.x;
  if (idx >= total4) return;
  const v4f xv = *(const v4f*)(x + (size_t)idx * 4);
  int c0 = (idx * 4) & (C - 1);   // C is a power of two
  v4h o;
#pragma unroll
  for (int j = 0; j < 4; ++j) {
    float r = xv[j] / s_a1[c0 + j];
    r = fminf(fmaxf(r, -128.f), 127.f);
    o[j] = (f16)rintf(r);
  }
  *(v4h*)(aq + (size_t)idx * 4) = o;
}

// ---------------------------------------------------------------------------
// Stage 2: statsq weight quant, one block per output row.
//   s = 2*mean(|w_row|)/sqrt(127);  q = clamp(round(w/s),-128,127)
//   outw[row][k] = q * s * colscale[k]
// Optional (fc2): cvec[row] = bias_row[row] + s * sum_k beta[k]*q
// ---------------------------------------------------------------------------
template <int ROWLEN>
__global__ __launch_bounds__(256) void k_quant_w(
    const float* __restrict__ w, const float* __restrict__ colscale,
    const float* __restrict__ beta, const float* __restrict__ bias_row,
    f16* __restrict__ outw, float* __restrict__ cvec) {
  __shared__ float red[256];
  const int row = blockIdx.x;
  const int tid = threadIdx.x;
  const float* wr = w + (size_t)row * ROWLEN;

  float sacc = 0.f;
#pragma unroll 4
  for (int i = tid; i < ROWLEN; i += 256) sacc += fabsf(wr[i]);
  red[tid] = sacc;
  __syncthreads();
#pragma unroll
  for (int off = 128; off > 0; off >>= 1) {
    if (tid < off) red[tid] += red[tid + off];
    __syncthreads();
  }
  const float s = 2.f * (red[0] * (1.f / (float)ROWLEN)) / SQRT127;
  __syncthreads();

  float zacc = 0.f;
#pragma unroll 4
  for (int i = tid; i < ROWLEN; i += 256) {
    float r = wr[i] / s;
    r = fminf(fmaxf(r, -128.f), 127.f);
    float q = rintf(r);
    outw[(size_t)row * ROWLEN + i] = (f16)(q * s * colscale[i]);
    if (beta) zacc += beta[i] * q;
  }
  if (cvec) {
    red[tid] = zacc;
    __syncthreads();
#pragma unroll
    for (int off = 128; off > 0; off >>= 1) {
      if (tid < off) red[tid] += red[tid + off];
      __syncthreads();
    }
    if (tid == 0) cvec[row] = bias_row[row] + s * red[0];
  }
}

// ---------------------------------------------------------------------------
// Stage 3: f16 WMMA GEMM, async double-buffered LDS pipeline.
// Block tile 256x128x64; 8 waves as 4(M) x 2(N); wave tile 64x64 = 4x4 WMMA accs.
// Per 32-k window: 8 fragment loads -> 16 WMMAs (1:1 ds:wmma).
// FUSE: epilogue = +b1, exact-erf GELU, asymmetric 8-bit quant -> f16 codes.
// else: epilogue = +cvec -> f32 out.
// ---------------------------------------------------------------------------
#define BM 256
#define BN 128
#define BK 64
#define LDT (BK + 8)  // padded LDS row stride (f16): 144 B, 16B-aligned

template <bool FUSE>
__global__ __launch_bounds__(256) void k_gemm(
    const f16* __restrict__ A, const f16* __restrict__ Bm,
    int M, int N, int K,
    const float* __restrict__ b1, const float* __restrict__ s_a2,
    const float* __restrict__ beta2, f16* __restrict__ out_f16,
    const float* __restrict__ cvec, float* __restrict__ out_f32) {
  constexpr int ASZ = BM * LDT;           // f16 elements
  constexpr int BUF = ASZ + BN * LDT;     // one double-buffer stage
  __shared__ f16 smem[2 * BUF];

  const int tid   = threadIdx.x;
  const int bM    = blockIdx.y * BM;
  const int bN    = blockIdx.x * BN;
  const int wave  = tid >> 5;
  const int lane  = tid & 31;
  const int lrow  = lane & 15;
  const int lhalf = lane >> 4;
  const int wM    = (wave & 3) * 64;   // 4 waves down M
  const int wN    = (wave >> 2) * 64;  // 2 waves across N

  // Copy coordinates: A = 2048 16B chunks (8/thread), B = 1024 (4/thread).
  const unsigned lds_base = (unsigned)(uintptr_t)&smem[0];
  unsigned aoff[8], ldsA[8], boff[4], ldsB[4];
#pragma unroll
  for (int it = 0; it < 8; ++it) {
    const int ch = tid + it * 256;
    const int r  = ch >> 3;
    const int c8 = (ch & 7) * 8;
    aoff[it] = (unsigned)(bM + r) * (unsigned)K + (unsigned)c8;
    ldsA[it] = lds_base + (unsigned)(r * LDT + c8) * 2u;
  }
#pragma unroll
  for (int it = 0; it < 4; ++it) {
    const int ch = tid + it * 256;
    const int r  = ch >> 3;
    const int c8 = (ch & 7) * 8;
    boff[it] = (unsigned)(bN + r) * (unsigned)K + (unsigned)c8;
    ldsB[it] = lds_base + (unsigned)(ASZ + r * LDT + c8) * 2u;
  }

  v8f acc[4][4] = {};

  // prologue: async-fill buffer 0 with the first K tile (12 ops/thread)
#pragma unroll
  for (int it = 0; it < 8; ++it) async_copy_b128(ldsA[it], A, aoff[it] * 2u);
#pragma unroll
  for (int it = 0; it < 4; ++it) async_copy_b128(ldsB[it], Bm, boff[it] * 2u);

  int buf = 0;
  for (int k0 = 0; k0 < K; k0 += BK) {
    const bool hasnext = (k0 + BK) < K;
    if (hasnext) {
      const unsigned bofs = (unsigned)((buf ^ 1) * BUF) * 2u;
      const unsigned kadd = (unsigned)(k0 + BK);
#pragma unroll
      for (int it = 0; it < 8; ++it)
        async_copy_b128(ldsA[it] + bofs, A, (aoff[it] + kadd) * 2u);
#pragma unroll
      for (int it = 0; it < 4; ++it)
        async_copy_b128(ldsB[it] + bofs, Bm, (boff[it] + kadd) * 2u);
      asm volatile("s_wait_asynccnt 0xc" ::: "memory");  // current tile done
    } else {
      asm volatile("s_wait_asynccnt 0x0" ::: "memory");  // drain
    }
    __syncthreads();  // all waves' portions of current tile visible

    const f16* As = &smem[buf * BUF];
    const f16* Bs = &smem[buf * BUF + ASZ];
#pragma unroll
    for (int ks = 0; ks < 2; ++ks) {  // two 32-wide K windows
      const int kb = ks * 32;
      v16h bf[4], af;
#pragma unroll
      for (int nt = 0; nt < 4; ++nt) {
        // B fragment: col = lrow, 16 contiguous k at (lane/16)*16
        const f16* p = &Bs[(wN + nt * 16 + lrow) * LDT + kb + lhalf * 16];
        v8h p0 = *(const v8h*)p;
        v8h p1 = *(const v8h*)(p + 8);
        bf[nt] = __builtin_shufflevector(p0, p1, 0, 1, 2, 3, 4, 5, 6, 7, 8, 9,
                                         10, 11, 12, 13, 14, 15);
      }
#pragma unroll
      for (int mt = 0; mt < 4; ++mt) {
        // A fragment: row = lrow, 16 contiguous k at (lane/16)*16
        const f16* p = &As[(wM + mt * 16 + lrow) * LDT + kb + lhalf * 16];
        v8h a0 = *(const v8h*)p;
        v8h a1 = *(const v8h*)(p + 8);
        af = __builtin_shufflevector(a0, a1, 0, 1, 2, 3, 4, 5, 6, 7, 8, 9, 10,
                                     11, 12, 13, 14, 15);
#pragma unroll
        for (int nt = 0; nt < 4; ++nt)
          acc[mt][nt] = __builtin_amdgcn_wmma_f32_16x16x32_f16(
              false, af, false, bf[nt], (short)0, acc[mt][nt], false, false);
      }
    }
    __syncthreads();  // all waves done reading this buffer -> safe to overwrite
    buf ^= 1;
  }

  // Epilogue.  C/D layout: element v of v8f holds row (v + 8*(lane/16)), col lane%16.
#pragma unroll
  for (int nt = 0; nt < 4; ++nt) {
    const int gc = bN + wN + nt * 16 + lrow;
    if (FUSE) {
      const float bias = b1[gc];
      const float sc   = s_a2[gc];
      const float bz   = beta2[gc];
#pragma unroll
      for (int mt = 0; mt < 4; ++mt) {
#pragma unroll
        for (int v = 0; v < 8; ++v) {
          const int gm = bM + wM + mt * 16 + lhalf * 8 + v;
          float h = acc[mt][nt][v] + bias;
          float g = 0.5f * h * (1.f + erff(h * 0.70710678118654752f));
          float r = (g - bz) / sc;
          r = fminf(fmaxf(r, 0.f), 255.f);
          out_f16[(size_t)gm * N + gc] = (f16)rintf(r);
        }
      }
    } else {
      const float bias = cvec[gc];
#pragma unroll
      for (int mt = 0; mt < 4; ++mt) {
#pragma unroll
        for (int v = 0; v < 8; ++v) {
          const int gm = bM + wM + mt * 16 + lhalf * 8 + v;
          out_f32[(size_t)gm * N + gc] = acc[mt][nt][v] + bias;
        }
      }
    }
  }
}

// ---------------------------------------------------------------------------
extern "C" void kernel_launch(void* const* d_in, const int* in_sizes, int n_in,
                              void* d_out, int out_size, void* d_ws,
                              size_t ws_size, hipStream_t stream) {
  const float* x     = (const float*)d_in[0];
  const float* w1    = (const float*)d_in[1];
  const float* b1    = (const float*)d_in[2];
  const float* w2    = (const float*)d_in[3];
  const float* b2    = (const float*)d_in[4];
  const float* s_a1  = (const float*)d_in[5];
  const float* s_a2  = (const float*)d_in[6];
  const float* beta2 = (const float*)d_in[7];
  float* out = (float*)d_out;

  const int C = 1024, H = 4096;
  const int M = 4 * 4096;  // B*N tokens

  char* ws = (char*)d_ws;
  f16* Aq1 = (f16*)ws;  ws += (size_t)M * C * sizeof(f16);   //  32 MB
  f16* Bq1 = (f16*)ws;  ws += (size_t)H * C * sizeof(f16);   //   8 MB
  f16* Aq2 = (f16*)ws;  ws += (size_t)M * H * sizeof(f16);   // 128 MB
  f16* Bq2 = (f16*)ws;  ws += (size_t)C * H * sizeof(f16);   //   8 MB
  float* cvec = (float*)ws;                                  //   4 KB

  // stage 1: quantize activations (int codes, f16)
  {
    int total4 = M * C / 4;
    k_quant_act<<<total4 / 256, 256, 0, stream>>>(x, s_a1, Aq1, total4, C);
  }
  // stage 2: statsq weights with scales folded; fc2 also folds zero-point bias
  k_quant_w<1024><<<H, 256, 0, stream>>>(w1, s_a1, nullptr, nullptr, Bq1, nullptr);
  k_quant_w<4096><<<C, 256, 0, stream>>>(w2, s_a2, beta2, b2, Bq2, cvec);

  // stage 3: fc1 GEMM + bias + exact GELU + asym quant (codes to Aq2)
  k_gemm<true><<<dim3(H / BN, M / BM), 256, 0, stream>>>(
      Aq1, Bq1, M, H, C, b1, s_a2, beta2, Aq2, nullptr, nullptr);
  // stage 4: fc2 GEMM + folded bias -> f32 out
  k_gemm<false><<<dim3(C / BN, M / BM), 256, 0, stream>>>(
      Aq2, Bq2, M, C, H, nullptr, nullptr, nullptr, nullptr, cvec, out);
}
